// GIN_22170621182212
// MI455X (gfx1250) — matrix operand order repaired and checked by reading, not compiled, not run on hardware
//
#include <hip/hip_runtime.h>
#include <hip/hip_bf16.h>

#define GIN_N 50000
#define GIN_D 128
#define GIN_E 800000
#define GIN_L 3

typedef __attribute__((ext_vector_type(16))) __bf16 v16bf;
typedef __attribute__((ext_vector_type(8)))  float  v8f;

// Fragment-major packed-weight geometry:
//   fragment = (tile(8) * 4 + kstep(4)) * 32 + lane(32), 16 bf16 each.
//   B element: n = tile*16 + (lane&15), k = kstep*32 + 16*(lane>>4) + e
#define FRAGS_PER_MAT (8 * 4 * 32)          // 1024 fragments
#define ELTS_PER_MAT  (FRAGS_PER_MAT * 16)  // 16384 bf16 per 128x128 matrix
#define N_MATS        (GIN_L * 2)           // 6 matrices total

// ---------------------------------------------------------------------------
// Kernel 0 (once per launch): pack all W1/W2 layers into fragment-major bf16.
// One thread per fragment: 16 strided f32 loads -> 16 bf16 -> one 32B store.
// ---------------------------------------------------------------------------
__global__ void gin_pack_weights_kernel(const float* __restrict__ W1,
                                        const float* __restrict__ W2,
                                        __bf16* __restrict__ wpack) {
  int f = blockIdx.x * blockDim.x + threadIdx.x;
  if (f >= N_MATS * FRAGS_PER_MAT) return;
  int lane  = f & 31;
  int kstep = (f >> 5) & 3;
  int tile  = (f >> 7) & 7;
  int mat   = f >> 10;                 // 0..5 = layer*2 + (0:W1, 1:W2)
  int layer = mat >> 1;
  const float* W = (mat & 1) ? (W2 + (size_t)layer * GIN_D * GIN_D)
                             : (W1 + (size_t)layer * GIN_D * GIN_D);
  int n     = tile * 16 + (lane & 15);
  int kbase = kstep * 32 + 16 * (lane >> 4);

  union { __bf16 h[16]; uint4 v[2]; } u;
#pragma unroll
  for (int e = 0; e < 16; ++e)
    u.h[e] = (__bf16)W[(size_t)(kbase + e) * GIN_D + n];

  uint4* dst = (uint4*)(wpack + (size_t)f * 16);
  dst[0] = u.v[0];
  dst[1] = u.v[1];
}

// ---------------------------------------------------------------------------
// Kernel 1: agg = x   (float4 copy)
// ---------------------------------------------------------------------------
__global__ void gin_copy_kernel(const float4* __restrict__ src,
                                float4* __restrict__ dst, int n4) {
  int i = blockIdx.x * blockDim.x + threadIdx.x;
  if (i < n4) dst[i] = src[i];
}

// ---------------------------------------------------------------------------
// Kernel 2: agg[dst[e]] += x[src[e]] — one wave32 per edge, lane = 4-float
// chunk (32 lanes * 4 floats = 128 = D). Edge id forced wave-uniform with
// readfirstlane so the index fetches scalarize to s_load (SMEM path),
// keeping VMEM free for the coalesced 512B row gather + f32 atomics.
// ---------------------------------------------------------------------------
__global__ void gin_scatter_kernel(const float* __restrict__ x,
                                   const int* __restrict__ esrc,
                                   const int* __restrict__ edst,
                                   float* __restrict__ agg) {
  int tid  = blockIdx.x * blockDim.x + threadIdx.x;
  int lane = tid & 31;
  int e    = __builtin_amdgcn_readfirstlane(tid >> 5);  // wave-uniform edge id
  if (e >= GIN_E) return;
  int s = esrc[e];
  int d = edst[e];
  const float4 v = *(const float4*)(x + (size_t)s * GIN_D + lane * 4);
  float* out = agg + (size_t)d * GIN_D + lane * 4;
  atomicAdd(out + 0, v.x);
  atomicAdd(out + 1, v.y);
  atomicAdd(out + 2, v.z);
  atomicAdd(out + 3, v.w);
}

// ---------------------------------------------------------------------------
// Kernel 3: Dout = relu(A @ W + bias).
// A:[N,128] f32; Wp: fragment-major bf16 (one matrix); bias:[128] f32.
// Block = 256 threads = 8 waves; block computes rows [16b, 16b+16) x all 128
// cols. A tile staged cooperatively into LDS as bf16 fragments (shared by all
// 8 waves); B fragments are single contiguous 32B loads from packed weights.
// 4x v_wmma_f32_16x16x32_bf16 per wave (K=128). In-place safe.
// ---------------------------------------------------------------------------
__global__ void gin_gemm_bias_relu_kernel(const float* __restrict__ A,
                                          const __bf16* __restrict__ Wp,
                                          const float* __restrict__ bias,
                                          float* __restrict__ Dout) {
  __shared__ __align__(32) __bf16 aTile[4 * 32 * 16];  // [kstep][lane][16] = 4KB

  const int t    = threadIdx.x;
  const int lane = t & 31;
  const int wave = t >> 5;
  const int half = lane >> 4;
  const int m16  = lane & 15;
  const int rowBase = blockIdx.x * 16;
  const int col     = wave * 16 + m16;

  // ---- cooperative A staging: thread t converts one 8-float run.
  //   fragment p = t>>1 = kstep*32 + flane ; sub = t&1 picks run 0/1.
  //   run floats are contiguous: A[rowBase+fm][kstep*32 + 16*fhalf + sub*8 ..+7]
  {
    int p     = t >> 1;
    int sub   = t & 1;
    int kstep = p >> 5;
    int fl    = p & 31;
    int fhalf = fl >> 4;
    int fm    = fl & 15;
    const float* src =
        A + (size_t)(rowBase + fm) * GIN_D + kstep * 32 + 16 * fhalf + sub * 8;
    union { __bf16 h[8]; uint4 v; } u;
#pragma unroll
    for (int e = 0; e < 8; ++e) u.h[e] = (__bf16)src[e];
    *(uint4*)(aTile + p * 16 + sub * 8) = u.v;
  }
  __syncthreads();

  // ---- C init = bias broadcast (column is constant across accumulator rows)
  float bval = bias[col];
  v8f c;
#pragma unroll
  for (int i = 0; i < 8; ++i) c[i] = bval;

  // ---- main loop: 4 K-steps of 32
  const __bf16* wbase = Wp + (size_t)(wave * 4) * 32 * 16;
#pragma unroll
  for (int kstep = 0; kstep < 4; ++kstep) {
    v16bf a = *(const v16bf*)(aTile + (kstep * 32 + lane) * 16);
    v16bf b = *(const v16bf*)(wbase + (size_t)(kstep * 32 + lane) * 16);
    c = __builtin_amdgcn_wmma_f32_16x16x32_bf16(
        false, a, false, b, (short)0, c, false, false);
  }

  // ---- store with ReLU: c[r] is element (m = r + 8*half, n = col)
#pragma unroll
  for (int r = 0; r < 8; ++r) {
    float val = fmaxf(c[r], 0.0f);
    Dout[(size_t)(rowBase + r + 8 * half) * GIN_D + col] = val;
  }
}

// ---------------------------------------------------------------------------
extern "C" void kernel_launch(void* const* d_in, const int* in_sizes, int n_in,
                              void* d_out, int out_size, void* d_ws, size_t ws_size,
                              hipStream_t stream) {
  const float* x0   = (const float*)d_in[0];   // [N, D]
  const int*   eidx = (const int*)d_in[1];     // [2, E] (int32)
  const float* W1   = (const float*)d_in[2];   // [L, D, D]
  const float* b1   = (const float*)d_in[3];   // [L, D]
  const float* W2   = (const float*)d_in[4];   // [L, D, D]
  const float* b2   = (const float*)d_in[5];   // [L, D]
  float*       out  = (float*)d_out;           // [N, D]

  const int* esrc = eidx;
  const int* edst = eidx + GIN_E;

  float*  xbuf   = (float*)d_ws;                     // N*D f32
  float*  aggbuf = xbuf + (size_t)GIN_N * GIN_D;     // N*D f32
  __bf16* wpack  = (__bf16*)(aggbuf + (size_t)GIN_N * GIN_D);  // 6*16384 bf16

  const int n4 = GIN_N * GIN_D / 4;
  const int copyBlocks    = (n4 + 255) / 256;
  const int scatterBlocks = (GIN_E * 32 + 255) / 256;
  const int gemmBlocks    = GIN_N / 16;               // 3125
  const int packBlocks    = (N_MATS * FRAGS_PER_MAT + 255) / 256;

  // one-time weight pre-pack (bf16, fragment-major)
  gin_pack_weights_kernel<<<packBlocks, 256, 0, stream>>>(W1, W2, wpack);

  const float* xcur = x0;
  for (int l = 0; l < GIN_L; ++l) {
    gin_copy_kernel<<<copyBlocks, 256, 0, stream>>>(
        (const float4*)xcur, (float4*)aggbuf, n4);
    gin_scatter_kernel<<<scatterBlocks, 256, 0, stream>>>(
        xcur, esrc, edst, aggbuf);
    // h = relu(agg @ W1[l] + b1[l])  (in-place on aggbuf)
    gin_gemm_bias_relu_kernel<<<gemmBlocks, 256, 0, stream>>>(
        aggbuf, wpack + (size_t)(l * 2 + 0) * ELTS_PER_MAT,
        b1 + (size_t)l * GIN_D, aggbuf);
    // x = relu(h @ W2[l] + b2[l])
    float* dst = (l == GIN_L - 1) ? out : xbuf;
    gin_gemm_bias_relu_kernel<<<gemmBlocks, 256, 0, stream>>>(
        aggbuf, wpack + (size_t)(l * 2 + 1) * ELTS_PER_MAT,
        b2 + (size_t)l * GIN_D, dst);
    xcur = dst;
  }
}
